// resourceAllocationNetwork_429496729976
// MI455X (gfx1250) — matrix operand order repaired and checked by reading, not compiled, not run on hardware
//
#include <hip/hip_runtime.h>
#include <hip/hip_bf16.h>

typedef __attribute__((ext_vector_type(16))) _Float16 v16h;
typedef __attribute__((ext_vector_type(8)))  _Float16 v8h;
typedef __attribute__((ext_vector_type(4)))  _Float16 v4h;
typedef __attribute__((ext_vector_type(8)))  float    v8f;

#define DEVINL __device__ __forceinline__

DEVINL v16h cat8(v8h lo, v8h hi) {
  return __builtin_shufflevector(lo, hi, 0,1,2,3,4,5,6,7,8,9,10,11,12,13,14,15);
}

// A-fragment (16x32 f16, MxK) in contiguous k order (B packed to match):
//   lanes 0-15 : M=lane,    K = {kb..kb+15} with kb = 32*kc
//   lanes 16-31: M=lane-16, same but kb += 16
// Activation tile is row-major [16][64] f16 in LDS; all chunk starts are
// multiples of 8 halves -> 16B-aligned ds_load_b128.
DEVINL v16h load_afrag(const _Float16* A, int lane, int kc) {
  const int row = lane & 15;
  const int kb  = kc * 32 + ((lane & 16) ? 16 : 0);
  v8h lo = *(const v8h*)(A + row * 64 + kb);
  v8h hi = *(const v8h*)(A + row * 64 + kb + 8);
  return cat8(lo, hi);
}

// B-fragment stored pre-swizzled in LDS: 32 lanes x 16 contiguous f16.
DEVINL v16h load_bfrag(const _Float16* p) {
  v8h lo = *(const v8h*)(p);
  v8h hi = *(const v8h*)(p + 8);
  return cat8(lo, hi);
}

// One layer for this wave's 16 rows: D = act(16xK) * W^T + bias, K padded to 64.
// NT output tiles of 16 features; C initialized to the bias (broadcast per lane).
template<int NT>
DEVINL void mm_layer(const _Float16* A, const _Float16* WF, const float* bias,
                     int outf, int lane, v8f* cout) {
  const v16h a0 = load_afrag(A, lane, 0);
  const v16h a1 = load_afrag(A, lane, 1);
#pragma unroll
  for (int t = 0; t < NT; ++t) {
    const int n = t * 16 + (lane & 15);
    const float bv = (n < outf) ? bias[n] : 0.0f;
    v8f c;
#pragma unroll
    for (int r = 0; r < 8; ++r) c[r] = bv;
    const v16h b0 = load_bfrag(WF + ((t * 2 + 0) * 32 + lane) * 16);
    c = __builtin_amdgcn_wmma_f32_16x16x32_f16(false, a0, false, b0,
                                               (short)0, c, false, false);
    const v16h b1 = load_bfrag(WF + ((t * 2 + 1) * 32 + lane) * 16);
    c = __builtin_amdgcn_wmma_f32_16x16x32_f16(false, a1, false, b1,
                                               (short)0, c, false, false);
    cout[t] = c;
  }
}

// D-matrix layout (ISA 7.12.2): lanes 0-15 -> N=lane, rows M=r; lanes 16-31 ->
// N=lane-16, rows M=r+8.  Write ReLU(f32)->f16 back to the row-major act tile.
// Note: for outf<48 this also re-zeroes the pad columns up to 47 every layer.
DEVINL void relu_store3(_Float16* A, const v8f* c, int row0, int colL) {
#pragma unroll
  for (int t = 0; t < 3; ++t)
#pragma unroll
    for (int r = 0; r < 8; ++r)
      A[(row0 + r) * 64 + t * 16 + colL] = (_Float16)fmaxf(c[t][r], 0.0f);
}

// Vectorized fill of the 40 data columns of the act tile from a global f32
// block with row stride 440: 160 aligned float4 chunks, 5 per lane.
DEVINL void fill40(_Float16* A, const float* __restrict__ xbase, long rowBase,
                   long maxRow, int lane) {
#pragma unroll
  for (int i = 0; i < 5; ++i) {
    const int cidx = i * 32 + lane;       // 0..159
    const int row  = cidx / 10;
    const int c4   = cidx - row * 10;     // float4 index within row
    const long gr  = min(rowBase + row, maxRow);
    const float4 f = *(const float4*)(xbase + gr * 440 + c4 * 4);
    v4h h;
    h[0] = (_Float16)f.x; h[1] = (_Float16)f.y;
    h[2] = (_Float16)f.z; h[3] = (_Float16)f.w;
    *(v4h*)(A + row * 64 + c4 * 4) = h;
  }
}

__launch_bounds__(256, 1)
__global__ void ran_fused_wmma_kernel(
    const float* __restrict__ x,
    const float* __restrict__ bW1, const float* __restrict__ bb1,
    const float* __restrict__ bW2, const float* __restrict__ bb2,
    const float* __restrict__ bW3, const float* __restrict__ bb3,
    const float* __restrict__ sW1, const float* __restrict__ sb1,
    const float* __restrict__ sW2, const float* __restrict__ sb2,
    const float* __restrict__ sW3, const float* __restrict__ sb3,
    float* __restrict__ out, int nrows)
{
  // 30 weight B-fragments (f16), 8 per-wave activation tiles, softmax scratch.
  __shared__ __align__(16) _Float16 wfrag[30][32][16];  // 30 KB
  __shared__ __align__(16) _Float16 act[8][16][64];     // 16 KB
  __shared__ __align__(16) float    smx[8][16][24];     // 12 KB
  __shared__ __align__(16) float    iabf[8][16][10];    //  5 KB

  const int tid  = threadIdx.x;
  const int wave = tid >> 5;
  const int lane = tid & 31;

  // ---- block-cooperative: convert f32 weights -> f16 WMMA B-fragments ----
  // B[k][n] = W[n][k]; lane n<16 holds W[n, kbase..kbase+15], lane n+16 holds
  // W[n, kbase+16..kbase+31]; zero-pad n>=outf or k>=inf.
  {
    const float* Ws[6] = {bW1, bW2, bW3, sW1, sW2, sW3};
    const int outfs[6] = {40, 40, 10, 41, 41, 22};
    const int infs[6]  = {40, 40, 40, 41, 41, 41};
    const int fbase[6] = {0, 6, 12, 14, 20, 26};
    for (int wi = tid; wi < 30 * 32; wi += 256) {
      const int frag = wi >> 5;
      const int l    = wi & 31;
      int layer = 5;
#pragma unroll
      for (int q = 4; q >= 0; --q)
        if (frag < fbase[q + 1]) layer = q;
      const int local = frag - fbase[layer];
      const int tile  = local >> 1;
      const int kc    = local & 1;
      const int n     = tile * 16 + (l & 15);
      const int kb    = kc * 32 + ((l & 16) ? 16 : 0);
      const float* W  = Ws[layer];
      const int of = outfs[layer], inf = infs[layer];
      _Float16* dst = &wfrag[frag][l][0];
#pragma unroll
      for (int j = 0; j < 16; ++j) {
        const int k = kb + j;
        const float v = (n < of && k < inf) ? W[n * inf + k] : 0.0f;
        dst[j] = (_Float16)v;
      }
    }
  }
  __syncthreads();

  const long rowBase = (long)blockIdx.x * 128 + wave * 16;
  const long maxRow  = (long)nrows - 1;
  _Float16* A  = &act[wave][0][0];
  float*    SM = &smx[wave][0][0];
  const int row0 = (lane >> 4) << 3;   // D-tile row offset for this lane
  const int colL = lane & 15;          // D-tile column for this lane
  v8f c[3];

  // ---- zero the pad columns 40..63 once (384 halves = 96 v4h chunks) ----
  {
    const v4h z = {};
#pragma unroll
    for (int i = 0; i < 3; ++i) {
      const int p   = i * 32 + lane;    // 0..95
      const int row = p / 6;
      const int c4  = p - row * 6;      // 6 chunks: cols 40..63
      *(v4h*)(A + row * 64 + 40 + c4 * 4) = z;
    }
  }

  // ---- big net input: x[:, 0:40] -> f16 ----
  fill40(A, x, rowBase, maxRow, lane);
  __builtin_amdgcn_wave_barrier();

  // big L1, L2 (40->40, ReLU)
  mm_layer<3>(A, &wfrag[0][0][0], bb1, 40, lane, c);
  __builtin_amdgcn_wave_barrier();
  relu_store3(A, c, row0, colL);
  __builtin_amdgcn_wave_barrier();
  mm_layer<3>(A, &wfrag[6][0][0], bb2, 40, lane, c);
  __builtin_amdgcn_wave_barrier();
  relu_store3(A, c, row0, colL);
  __builtin_amdgcn_wave_barrier();

  // big L3 (40->10): logits to f32 scratch
  mm_layer<1>(A, &wfrag[12][0][0], bb3, 10, lane, c);
#pragma unroll
  for (int r = 0; r < 8; ++r)
    if (colL < 10) SM[(row0 + r) * 24 + colL] = c[0][r];
  __builtin_amdgcn_wave_barrier();

  // softmax over 10 -> iab, one row per lane (lanes 0..15)
  if (lane < 16) {
    const float* rowp = SM + lane * 24;
    float m = rowp[0];
#pragma unroll
    for (int j = 1; j < 10; ++j) m = fmaxf(m, rowp[j]);
    float e[10], ssum = 0.0f;
#pragma unroll
    for (int j = 0; j < 10; ++j) { e[j] = __expf(rowp[j] - m); ssum += e[j]; }
    const float inv = 1.0f / ssum;
#pragma unroll
    for (int j = 0; j < 10; ++j) iabf[wave][lane][j] = e[j] * inv;
  }
  __builtin_amdgcn_wave_barrier();

  // ---- 10 small nets ----
  for (int s = 0; s < 10; ++s) {
    if (s < 9)
      __builtin_prefetch(x + rowBase * 440 + 40 + (s + 1) * 40, 0, 1);

    // input cols 0..39 = x_ues[:, s, :]; col 40 = iab; 41..63 already zero
    fill40(A, x + 40 + s * 40, rowBase, maxRow, lane);
    if (lane < 16)
      A[lane * 64 + 40] = (_Float16)iabf[wave][lane][s];
    __builtin_amdgcn_wave_barrier();

    mm_layer<3>(A, &wfrag[14][0][0], sb1, 41, lane, c);
    __builtin_amdgcn_wave_barrier();
    relu_store3(A, c, row0, colL);
    __builtin_amdgcn_wave_barrier();
    mm_layer<3>(A, &wfrag[20][0][0], sb2, 41, lane, c);
    __builtin_amdgcn_wave_barrier();
    relu_store3(A, c, row0, colL);
    __builtin_amdgcn_wave_barrier();

    mm_layer<2>(A, &wfrag[26][0][0], sb3, 22, lane, c);
#pragma unroll
    for (int t = 0; t < 2; ++t)
#pragma unroll
      for (int r = 0; r < 8; ++r)
        if (t * 16 + colL < 22) SM[(row0 + r) * 24 + t * 16 + colL] = c[t][r];
    __builtin_amdgcn_wave_barrier();

    // softmax over 22, scale by iab, write out[b, s, :] as float2 pairs
    if (lane < 16 && rowBase + lane < nrows) {
      const float* rowp = SM + lane * 24;
      float m = rowp[0];
#pragma unroll
      for (int j = 1; j < 22; ++j) m = fmaxf(m, rowp[j]);
      float e[22], ssum = 0.0f;
#pragma unroll
      for (int j = 0; j < 22; ++j) { e[j] = __expf(rowp[j] - m); ssum += e[j]; }
      const float scale = iabf[wave][lane][s] / ssum;
      float2* op = (float2*)(out + ((rowBase + lane) * 10 + s) * 22);
#pragma unroll
      for (int j = 0; j < 11; ++j) {
        float2 p; p.x = e[2 * j] * scale; p.y = e[2 * j + 1] * scale;
        op[j] = p;
      }
    }
    __builtin_amdgcn_wave_barrier();
  }
}

extern "C" void kernel_launch(void* const* d_in, const int* in_sizes, int n_in,
                              void* d_out, int out_size, void* d_ws, size_t ws_size,
                              hipStream_t stream) {
  (void)n_in; (void)out_size; (void)d_ws; (void)ws_size;
  const float* x   = (const float*)d_in[0];
  const float* bW1 = (const float*)d_in[1];
  const float* bb1 = (const float*)d_in[2];
  const float* bW2 = (const float*)d_in[3];
  const float* bb2 = (const float*)d_in[4];
  const float* bW3 = (const float*)d_in[5];
  const float* bb3 = (const float*)d_in[6];
  const float* sW1 = (const float*)d_in[7];
  const float* sb1 = (const float*)d_in[8];
  const float* sW2 = (const float*)d_in[9];
  const float* sb2 = (const float*)d_in[10];
  const float* sW3 = (const float*)d_in[11];
  const float* sb3 = (const float*)d_in[12];
  float* out = (float*)d_out;

  const int nrows  = in_sizes[0] / 440;           // 262144
  const int blocks = (nrows + 127) / 128;         // 128 rows per 256-thread block
  ran_fused_wmma_kernel<<<blocks, 256, 0, stream>>>(
      x, bW1, bb1, bW2, bb2, bW3, bb3, sW1, sb1, sW2, sb2, sW3, sb3,
      out, nrows);
}
